// DifferentialAttention_32349693673870
// MI455X (gfx1250) — matrix operand order repaired and checked by reading, not compiled, not run on hardware
//
#include <hip/hip_runtime.h>

// ---------------------------------------------------------------------------
// Differential attention, CDNA5 (gfx1250), wave32 + v_wmma_f32_16x16x32_bf16
//   B=2, S=2048, D=2048, H=16, hd=128, qd=64, scale = 128^-0.5
//   Attention kernel double-buffers K/V blocks in LDS via
//   global_load_async_to_lds_b128 (ASYNCcnt) for latency hiding.
// ---------------------------------------------------------------------------

typedef __attribute__((ext_vector_type(16))) __bf16 v16bf;
typedef __attribute__((ext_vector_type(8)))  __bf16 v8bf;
typedef __attribute__((ext_vector_type(8)))  float  v8f;

static __device__ __forceinline__ v8f wmma_bf16(v16bf a, v16bf b, v8f c) {
  // (neg_a, A, neg_b, B, c_mod, C, reuse_a, reuse_b)
  return __builtin_amdgcn_wmma_f32_16x16x32_bf16(false, a, false, b,
                                                 (short)0, c, false, false);
}

// Build a v16bf A/B fragment from two 16-byte chunks at +0 and +32 bytes.
static __device__ __forceinline__ v16bf mk16(const v8bf* p) {
  v8bf lo = p[0];
  v8bf hi = p[2];   // +16 elements = +32 bytes
  v16bf r;
#pragma unroll
  for (int i = 0; i < 8; ++i) { r[i] = lo[i]; r[i + 8] = hi[i]; }
  return r;
}

// Per-lane async copy of 16 bytes global -> LDS (tracked by ASYNCcnt).
#define ASYNC_CP16(ldsOff, gAddr)                                        \
  asm volatile("global_load_async_to_lds_b128 %0, %1, off"               \
               :: "v"(ldsOff), "v"(gAddr) : "memory")

// ------------------------- conversion kernels ------------------------------

__global__ void k_convert_x(const float* __restrict__ x,
                            __bf16* __restrict__ xb, int n) {
  int i = blockIdx.x * 256 + threadIdx.x;
  if (i < n) xb[i] = (__bf16)x[i];
}

// w: [rows][cols] fp32  ->  wt: [cols][rows] bf16   (wt row n holds K contig)
__global__ void k_transpose_bf16(const float* __restrict__ w,
                                 __bf16* __restrict__ wt,
                                 int rows, int cols) {
  int i = blockIdx.x * 256 + threadIdx.x;
  int total = rows * cols;
  if (i < total) {
    int n = i / rows;       // column of w = row of wt
    int k = i % rows;       // row of w   = col of wt
    wt[i] = (__bf16)w[(long long)k * cols + n];
  }
}

// ------------------------- QKV GEMM (4096x2048 @ 2048x6144) ----------------
// One wave computes a 64x64 tile; scatters into attention layouts.

__global__ void __launch_bounds__(256) k_qkv_gemm(
    const __bf16* __restrict__ Xb,    // [4096][2048]
    const __bf16* __restrict__ WaT,   // [6144][2048]  (transposed weights)
    const float*  __restrict__ bias,  // [6144]
    __bf16* __restrict__ q1, __bf16* __restrict__ q2,
    __bf16* __restrict__ k1, __bf16* __restrict__ k2,
    __bf16* __restrict__ vT)          // [B][H][128][2048]
{
  const int K = 2048, NTILES = 6144 / 64;           // 96
  int wave = blockIdx.x * 8 + (threadIdx.x >> 5);   // 6144 waves total
  int lane = threadIdx.x & 31;
  int mt = wave / NTILES, nt = wave % NTILES;
  int m0 = mt * 64, n0 = nt * 64;
  int colIdx  = lane & 15;
  int halfSel = (lane < 16) ? 0 : 1;
  int ca = halfSel ? 8 : 0;        // A-operand K-chunk base
  int cb = halfSel ? 16 : 0;       // unused; B operand also uses ca

  v8f acc[4][4];
#pragma unroll
  for (int j = 0; j < 4; ++j) {
    float bv = bias[n0 + j * 16 + colIdx];           // per-column bias
#pragma unroll
    for (int i = 0; i < 4; ++i)
#pragma unroll
      for (int r = 0; r < 8; ++r) acc[i][j][r] = bv;
  }

  for (int kb = 0; kb < K; kb += 32) {
    v16bf a[4], b[4];
#pragma unroll
    for (int i = 0; i < 4; ++i)
      a[i] = mk16((const v8bf*)(Xb + (long long)(m0 + i * 16 + colIdx) * K + kb + ca));
#pragma unroll
    for (int j = 0; j < 4; ++j)
      b[j] = mk16((const v8bf*)(WaT + (long long)(n0 + j * 16 + colIdx) * K + kb + ca));
    if (kb + 32 < K)  // warm next B slab (global_prefetch_b8)
      __builtin_prefetch(WaT + (long long)(n0 + colIdx) * K + kb + 32, 0, 0);
#pragma unroll
    for (int i = 0; i < 4; ++i)
#pragma unroll
      for (int j = 0; j < 4; ++j)
        acc[i][j] = wmma_bf16(a[i], b[j], acc[i][j]);
  }

  // Scatter C into q1/q2/k1/k2 [B][H][S][64] and vT [B][H][128][S]
#pragma unroll
  for (int i = 0; i < 4; ++i) {
    int rowBase = m0 + i * 16 + (halfSel ? 8 : 0);
#pragma unroll
    for (int j = 0; j < 4; ++j) {
      int col = n0 + j * 16 + colIdx;
#pragma unroll
      for (int r = 0; r < 8; ++r) {
        int row = rowBase + r;
        int bi = row >> 11, s = row & 2047;
        __bf16 val = (__bf16)acc[i][j][r];
        if (col < 1024) {
          int h = col >> 6, d = col & 63;
          q1[(((long long)(bi * 16 + h)) * 2048 + s) * 64 + d] = val;
        } else if (col < 2048) {
          int c = col - 1024; int h = c >> 6, d = c & 63;
          q2[(((long long)(bi * 16 + h)) * 2048 + s) * 64 + d] = val;
        } else if (col < 3072) {
          int c = col - 2048; int h = c >> 6, d = c & 63;
          k1[(((long long)(bi * 16 + h)) * 2048 + s) * 64 + d] = val;
        } else if (col < 4096) {
          int c = col - 3072; int h = c >> 6, d = c & 63;
          k2[(((long long)(bi * 16 + h)) * 2048 + s) * 64 + d] = val;
        } else {
          int c = col - 4096; int h = c >> 7, d = c & 127;
          vT[(((long long)(bi * 16 + h)) * 128 + d) * 2048 + s] = val;
        }
      }
    }
  }
}

// ------------------ differential flash attention ---------------------------
// grid (S/16, B*H), block 32 (one wave). 16 query rows, stream keys by 32.
// K/V blocks double-buffered in LDS via async copies (ASYNCcnt pipeline).

__global__ void __launch_bounds__(32) k_diff_attn(
    const __bf16* __restrict__ q1b, const __bf16* __restrict__ q2b,
    const __bf16* __restrict__ k1b, const __bf16* __restrict__ k2b,
    const __bf16* __restrict__ vT,
    const float* __restrict__ amask, const float* __restrict__ lam_p,
    __bf16* __restrict__ attnOut)   // [4096][2048] bf16
{
  const int S = 2048;
  int qt = blockIdx.x;
  int bh = blockIdx.y;
  int b = bh >> 4, h = bh & 15;
  int lane = threadIdx.x;
  int colIdx = lane & 15;
  int half8 = (lane >> 4) * 8;
  int ca = (lane < 16) ? 0 : 8;    // A-operand chunk
  int cb = (lane < 16) ? 0 : 16;   // unused; B operands also use ca
  int q0 = qt * 16;

  __shared__ alignas(16) float  sc[2][16][32];   // scores (fp32)
  __shared__ alignas(16) __bf16 pb[2][16][32];   // probabilities (bf16)
  __shared__ float mst[2][16], lst[2][16], fsc[2][16], mbuf[32];
  // double-buffered async staging:  K: [buf][branch][key 32][d 64]
  //                                 V: [buf][d 128][key 32]
  __shared__ alignas(32) __bf16 kst[2][2][32][64];   // 16 KB
  __shared__ alignas(32) __bf16 vst[2][128][32];     // 16 KB

  const __bf16* qsrc[2] = { q1b, q2b };
  const __bf16* ksrc[2] = { k1b, k2b };
  long long headOff = ((long long)(b * 16 + h)) * 2048 * 64;
  long long vHead   = ((long long)(b * 16 + h)) * 128 * 2048;

  // Issue one block's async staging: 16 (K) + 16 (V) = 32 async instructions.
  auto prefetch = [&](int sel, int kb) {
    #pragma unroll
    for (int br = 0; br < 2; ++br) {
      unsigned long long src = (unsigned long long)(uintptr_t)
          (ksrc[br] + headOff + (long long)(kb + lane) * 64);
      unsigned dst = (unsigned)(uintptr_t)&kst[sel][br][lane][0];
      #pragma unroll
      for (int c = 0; c < 8; ++c)
        ASYNC_CP16(dst + (unsigned)(c * 16), src + (unsigned long long)(c * 16));
    }
    #pragma unroll
    for (int r = 0; r < 4; ++r) {
      int d = r * 32 + lane;
      unsigned long long src = (unsigned long long)(uintptr_t)
          (vT + vHead + (long long)d * 2048 + kb);
      unsigned dst = (unsigned)(uintptr_t)&vst[sel][d][0];
      #pragma unroll
      for (int c = 0; c < 4; ++c)
        ASYNC_CP16(dst + (unsigned)(c * 16), src + (unsigned long long)(c * 16));
    }
  };

  // Preload Q fragments (K = 64 -> 2 steps per branch)
  v16bf qf[2][2];
#pragma unroll
  for (int br = 0; br < 2; ++br)
#pragma unroll
    for (int ks = 0; ks < 2; ++ks)
      qf[br][ks] = mk16((const v8bf*)(qsrc[br] + headOff +
                        (long long)(q0 + colIdx) * 64 + ks * 32 + ca));

  { int t = lane & 15, br = lane >> 4;
    mst[br][t] = -3.0e38f; lst[br][t] = 0.0f; }
  __syncthreads();

  v8f O[2][8];
#pragma unroll
  for (int br = 0; br < 2; ++br)
#pragma unroll
    for (int j = 0; j < 8; ++j) O[br][j] = (v8f){};

  const float scale = 0.08838834764831845f;  // 128^-0.5

  prefetch(0, 0);   // fill first buffer

  for (int kb = 0; kb < S; kb += 32) {
    int sel = (kb >> 5) & 1;
    if (kb + 32 < S) {
      prefetch(sel ^ 1, kb + 32);              // 32 async ops in flight for next
      asm volatile("s_wait_asynccnt 0x20" ::: "memory");  // current block done
    } else {
      asm volatile("s_wait_asynccnt 0x0" ::: "memory");
    }

    mbuf[lane] = (1.0f - amask[b * S + kb + lane]) * -10000.0f;

    // --- scores: S_br = Q_br @ K_br^T  (2 N-subtiles x 2 K-steps each) ---
#pragma unroll
    for (int br = 0; br < 2; ++br) {
#pragma unroll
      for (int jt = 0; jt < 2; ++jt) {
        v8f s = (v8f){};
#pragma unroll
        for (int ks = 0; ks < 2; ++ks) {
          v16bf kf = mk16((const v8bf*)&kst[sel][br][jt * 16 + colIdx][ks * 32 + ca]);
          s = wmma_bf16(qf[br][ks], kf, s);
        }
#pragma unroll
        for (int r = 0; r < 8; ++r)
          sc[br][half8 + r][jt * 16 + colIdx] = s[r] * scale;
      }
    }
    __syncthreads();

    // --- per-row online softmax (lanes 0-15: branch 0, 16-31: branch 1) ---
    {
      int t = lane & 15, br = lane >> 4;
      float mo = mst[br][t], lo = lst[br][t];
      float rm = -3.0e38f;
#pragma unroll
      for (int k = 0; k < 32; ++k) {
        float v = sc[br][t][k] + mbuf[k];
        sc[br][t][k] = v;
        rm = fmaxf(rm, v);
      }
      float mn = fmaxf(mo, rm);
      float f  = __expf(mo - mn);
      float sum = 0.0f;
#pragma unroll
      for (int k = 0; k < 32; ++k) {
        float p = __expf(sc[br][t][k] - mn);
        pb[br][t][k] = (__bf16)p;
        sum += p;
      }
      mst[br][t] = mn;
      lst[br][t] = lo * f + sum;
      fsc[br][t] = f;
    }
    __syncthreads();

    // --- rescale accumulators, build P A-fragments ---
    v16bf pf[2];
#pragma unroll
    for (int br = 0; br < 2; ++br) {
      float fr[8];
#pragma unroll
      for (int r = 0; r < 8; ++r) fr[r] = fsc[br][half8 + r];
#pragma unroll
      for (int j = 0; j < 8; ++j)
#pragma unroll
        for (int r = 0; r < 8; ++r) O[br][j][r] *= fr[r];
      pf[br] = mk16((const v8bf*)(&pb[br][colIdx][ca]));
    }

    // --- O_br += P_br @ V : 8 N-subtiles, V fragment shared by branches ---
#pragma unroll
    for (int jt = 0; jt < 8; ++jt) {
      v16bf vf = mk16((const v8bf*)&vst[sel][jt * 16 + colIdx][ca]);
      O[0][jt] = wmma_bf16(pf[0], vf, O[0][jt]);
      O[1][jt] = wmma_bf16(pf[1], vf, O[1][jt]);
    }
    __syncthreads();
  }

  // --- finalize: out = O1/l1 - lambda * O2/l2 ---
  float lam = lam_p[0];
  float il1[8], il2[8];
#pragma unroll
  for (int r = 0; r < 8; ++r) {
    il1[r] = 1.0f / lst[0][half8 + r];
    il2[r] = 1.0f / lst[1][half8 + r];
  }
#pragma unroll
  for (int jt = 0; jt < 8; ++jt)
#pragma unroll
    for (int r = 0; r < 8; ++r) {
      float o = O[0][jt][r] * il1[r] - lam * O[1][jt][r] * il2[r];
      int srow = q0 + half8 + r;
      attnOut[(long long)(b * 2048 + srow) * 2048 + h * 128 + jt * 16 + colIdx]
          = (__bf16)o;
    }
}

// ------------------- output projection (4096x2048 @ 2048x2048) -------------

__global__ void __launch_bounds__(256) k_proj_gemm(
    const __bf16* __restrict__ Ab,    // [4096][2048] bf16
    const __bf16* __restrict__ WpT,   // [2048][2048] transposed
    const float*  __restrict__ bias,  // [2048]
    float* __restrict__ out)          // [4096][2048] fp32
{
  const int K = 2048, NTILES = 2048 / 64;           // 32
  int wave = blockIdx.x * 8 + (threadIdx.x >> 5);   // 2048 waves
  int lane = threadIdx.x & 31;
  int mt = wave / NTILES, nt = wave % NTILES;
  int m0 = mt * 64, n0 = nt * 64;
  int colIdx  = lane & 15;
  int halfSel = (lane < 16) ? 0 : 1;
  int ca = halfSel ? 8 : 0;
  int cb = halfSel ? 16 : 0;

  v8f acc[4][4];
#pragma unroll
  for (int j = 0; j < 4; ++j) {
    float bv = bias[n0 + j * 16 + colIdx];
#pragma unroll
    for (int i = 0; i < 4; ++i)
#pragma unroll
      for (int r = 0; r < 8; ++r) acc[i][j][r] = bv;
  }

  for (int kb = 0; kb < K; kb += 32) {
    v16bf a[4], b[4];
#pragma unroll
    for (int i = 0; i < 4; ++i)
      a[i] = mk16((const v8bf*)(Ab + (long long)(m0 + i * 16 + colIdx) * K + kb + ca));
#pragma unroll
    for (int j = 0; j < 4; ++j)
      b[j] = mk16((const v8bf*)(WpT + (long long)(n0 + j * 16 + colIdx) * K + kb + ca));
    if (kb + 32 < K)
      __builtin_prefetch(WpT + (long long)(n0 + colIdx) * K + kb + 32, 0, 0);
#pragma unroll
    for (int i = 0; i < 4; ++i)
#pragma unroll
      for (int j = 0; j < 4; ++j)
        acc[i][j] = wmma_bf16(a[i], b[j], acc[i][j]);
  }

#pragma unroll
  for (int i = 0; i < 4; ++i) {
    int rowBase = m0 + i * 16 + (halfSel ? 8 : 0);
#pragma unroll
    for (int j = 0; j < 4; ++j) {
      int col = n0 + j * 16 + colIdx;
#pragma unroll
      for (int r = 0; r < 8; ++r)
        out[(long long)(rowBase + r) * 2048 + col] = acc[i][j][r];
    }
  }
}

// ------------------------------- launch ------------------------------------

extern "C" void kernel_launch(void* const* d_in, const int* in_sizes, int n_in,
                              void* d_out, int out_size, void* d_ws, size_t ws_size,
                              hipStream_t stream) {
  const float* hs  = (const float*)d_in[0];   // hidden_states [2,2048,2048]
  const float* am  = (const float*)d_in[1];   // attention_mask [2,2048]
  const float* Wa  = (const float*)d_in[2];   // W_attn [2048,6144]
  const float* ba  = (const float*)d_in[3];   // b_attn [6144]
  const float* Wp  = (const float*)d_in[4];   // W_proj [2048,2048]
  const float* bp  = (const float*)d_in[5];   // b_proj [2048]
  const float* lam = (const float*)d_in[6];   // lambda_param [1]

  char* ws = (char*)d_ws;
  __bf16* Xb   = (__bf16*)(ws);                 // 16 MB  [4096][2048]
  __bf16* WaT  = (__bf16*)(ws + 16777216);      // 24 MB  [6144][2048]
  __bf16* WpT  = (__bf16*)(ws + 41943040);      //  8 MB  [2048][2048]
  __bf16* q1   = (__bf16*)(ws + 50331648);      //  8 MB  [2][16][2048][64]
  __bf16* q2   = (__bf16*)(ws + 58720256);
  __bf16* k1   = (__bf16*)(ws + 67108864);
  __bf16* k2   = (__bf16*)(ws + 75497472);
  __bf16* vT   = (__bf16*)(ws + 83886080);      // 16 MB  [2][16][128][2048]
  __bf16* attn = (__bf16*)(ws + 100663296);     // 16 MB  [4096][2048]

  k_convert_x<<<8388608 / 256, 256, 0, stream>>>(hs, Xb, 8388608);
  k_transpose_bf16<<<12582912 / 256, 256, 0, stream>>>(Wa, WaT, 2048, 6144);
  k_transpose_bf16<<<4194304 / 256, 256, 0, stream>>>(Wp, WpT, 2048, 2048);

  k_qkv_gemm<<<768, 256, 0, stream>>>(Xb, WaT, ba, q1, q2, k1, k2, vT);

  dim3 g2(128, 32);   // (S/16 query tiles, B*H heads)
  k_diff_attn<<<g2, 32, 0, stream>>>(q1, q2, k1, k2, vT, am, lam, attn);

  k_proj_gemm<<<256, 256, 0, stream>>>(attn, WpT, bp, (float*)d_out);
}
